// Mamba3MIMOBlock_24653112279098
// MI455X (gfx1250) — matrix operand order repaired and checked by reading, not compiled, not run on hardware
//
#include <hip/hip_runtime.h>
#include <stdint.h>

#define D_MODEL 2048
#define RANK    64
#define NCAT    192            // 3 * RANK: [W_dt; W_ph; W_B]
#define BATCH   16384
#define ROWS    64             // rows of x per workgroup
#define NWAVES  12
#define NTHREADS (NWAVES * 32)

typedef __attribute__((ext_vector_type(16))) __bf16 v16bf;
typedef __attribute__((ext_vector_type(8)))  float  v8f;

union BF16Frag { uint32_t u[8]; v16bf v; };

// ---- bf16 helpers: round-to-nearest-even truncation + hi/lo split -------
__device__ __forceinline__ uint16_t f2bf(float f) {
  uint32_t u = __builtin_bit_cast(uint32_t, f);
  u += 0x7FFFu + ((u >> 16) & 1u);
  return (uint16_t)(u >> 16);
}
__device__ __forceinline__ float bf2f(uint16_t h) {
  uint32_t u = ((uint32_t)h) << 16;
  return __builtin_bit_cast(float, u);
}

__device__ __forceinline__ v8f wmma_bf16(const BF16Frag& A, const BF16Frag& B, v8f C) {
  // D = A(16x32 bf16) x B(32x16 bf16) + C(16x16 f32)
  return __builtin_amdgcn_wmma_f32_16x16x32_bf16(
      /*neg_a=*/false, A.v, /*neg_b=*/false, B.v,
      /*c_mod=*/(short)0, C, /*reuse_a=*/false, /*reuse_b=*/false);
}

// A fragment (16x32, rows rowBase..+15, K window k0..k0+31) from LDS tile
// stored row-major [..][ld] (ld multiple of 8 -> 16B-aligned b128 loads).
// ISA layout: lanes 0-15 M=0..15; v0..3: K=0..7(+8*grp), v4..7: K=16..23(+8*grp)
__device__ __forceinline__ void load_A_lds(const uint16_t* base, int ld,
                                           int rowBase, int k0, int lane,
                                           BF16Frag& F) {
  int r = lane & 15, grp = lane >> 4;
  const uint16_t* p = base + (rowBase + r) * ld + k0 + 8 * grp;
  uint4 a0 = *(const uint4*)p;          // ds_load_b128
  uint4 a1 = *(const uint4*)(p + 16);   // ds_load_b128
  F.u[0] = a0.x; F.u[1] = a0.y; F.u[2] = a0.z; F.u[3] = a0.w;
  F.u[4] = a1.x; F.u[5] = a1.y; F.u[6] = a1.z; F.u[7] = a1.w;
}

// B fragment (32x16, cols n, K window k0..k0+31) from global row-major W[n][ld].
// ISA layout: v0..3: K=0..7(+8*grp), v4..7: K=16..23(+8*grp).
__device__ __forceinline__ void load_B_glb(const uint16_t* __restrict__ W, int n,
                                           int ld, int k0, int lane, BF16Frag& F) {
  int grp = lane >> 4;
  const uint16_t* p = W + (size_t)n * ld + k0 + 8 * grp;
  uint4 b0 = *(const uint4*)p;          // global_load_b128
  uint4 b1 = *(const uint4*)(p + 16);   // global_load_b128
  F.u[0] = b0.x; F.u[1] = b0.y; F.u[2] = b0.z; F.u[3] = b0.w;
  F.u[4] = b1.x; F.u[5] = b1.y; F.u[6] = b1.z; F.u[7] = b1.w;
}

// ---- weight pre-split: f32 -> bf16 hi + bf16 residual (one-time, 2 MB) ---
__global__ void prep_weights(const float* __restrict__ Wdt,
                             const float* __restrict__ Wph,
                             const float* __restrict__ WB,
                             const float* __restrict__ WC,
                             uint16_t* __restrict__ wcat_hi,
                             uint16_t* __restrict__ wcat_lo,
                             uint16_t* __restrict__ wc_hi,
                             uint16_t* __restrict__ wc_lo) {
  int idx = blockIdx.x * blockDim.x + threadIdx.x;
  if (idx < NCAT * D_MODEL) {
    int row = idx / D_MODEL, k = idx - row * D_MODEL;
    float v;
    if (row < RANK)           v = Wdt[row * D_MODEL + k];
    else if (row < 2 * RANK)  v = Wph[(row - RANK) * D_MODEL + k];
    else                      v = WB[(row - 2 * RANK) * D_MODEL + k];
    uint16_t hi = f2bf(v);
    wcat_hi[idx] = hi;
    wcat_lo[idx] = f2bf(v - bf2f(hi));
  }
  if (idx < D_MODEL * RANK) {           // W_C is already [n=2048][k=64]
    float v = WC[idx];
    uint16_t hi = f2bf(v);
    wc_hi[idx] = hi;
    wc_lo[idx] = f2bf(v - bf2f(hi));
  }
}

// ---- fused kernel: GEMM1 -> state update -> GEMM2 ------------------------
__global__ __launch_bounds__(NTHREADS)
void mamba3_fused(const float* __restrict__ x,
                  const float* __restrict__ state_re,
                  const float* __restrict__ state_im,
                  const float* __restrict__ b_dt,
                  const float* __restrict__ b_ph,
                  const float* __restrict__ A_real,
                  const float* __restrict__ A_imag,
                  const uint16_t* __restrict__ wcat_hi,
                  const uint16_t* __restrict__ wcat_lo,
                  const uint16_t* __restrict__ wc_hi,
                  const uint16_t* __restrict__ wc_lo,
                  float* __restrict__ out,
                  float* __restrict__ new_re,
                  float* __restrict__ new_im) {
  __shared__ uint16_t Xhi[ROWS][64];          // 8 KB  (x chunk, bf16 hi)
  __shared__ uint16_t Xlo[ROWS][64];          // 8 KB  (x chunk, bf16 residual)
  __shared__ float    raw[3][ROWS][RANK];     // 48 KB (dt / phase / B_val pre-act)
  __shared__ uint16_t Rhi[ROWS][RANK];        // 8 KB  (new_re hi)
  __shared__ uint16_t Rlo[ROWS][RANK];        // 8 KB  (new_re residual)

  const int tid   = threadIdx.x;
  const int lane  = tid & 31;
  const int wave  = tid >> 5;                 // 0..11 == N-tile of GEMM1
  const int rowG0 = blockIdx.x * ROWS;

  // ================= GEMM1: [64 x 2048] x [2048 x 192] =================
  v8f zero = {};
  v8f acc[4] = {zero, zero, zero, zero};      // 4 M-tiles per wave

  for (int kc = 0; kc < D_MODEL; kc += 64) {
    __syncthreads();                          // protect LDS chunk reuse
    // stage x chunk: float4 loads, packed bf16 hi/lo, ds_store_b64
    for (int i = tid; i < ROWS * 16; i += NTHREADS) {
      int r = i >> 4, k4 = (i & 15) << 2;
      const float4 v4 =
          *(const float4*)&x[(size_t)(rowG0 + r) * D_MODEL + kc + k4];
      uint16_t h0 = f2bf(v4.x), h1 = f2bf(v4.y);
      uint16_t h2 = f2bf(v4.z), h3 = f2bf(v4.w);
      uint16_t l0 = f2bf(v4.x - bf2f(h0)), l1 = f2bf(v4.y - bf2f(h1));
      uint16_t l2 = f2bf(v4.z - bf2f(h2)), l3 = f2bf(v4.w - bf2f(h3));
      uint2 hp, lp;
      hp.x = (uint32_t)h0 | ((uint32_t)h1 << 16);
      hp.y = (uint32_t)h2 | ((uint32_t)h3 << 16);
      lp.x = (uint32_t)l0 | ((uint32_t)l1 << 16);
      lp.y = (uint32_t)l2 | ((uint32_t)l3 << 16);
      *(uint2*)&Xhi[r][k4] = hp;
      *(uint2*)&Xlo[r][k4] = lp;
    }
    __syncthreads();

    // prefetch NEXT x chunk into cache (global_prefetch_b8: no LOADcnt,
    // so it cannot stall the WMMA B-load waits below). 128B per thread,
    // threads 0..127 cover 64 rows x 256B.
    if (kc + 64 < D_MODEL && tid < 128) {
      int r = tid >> 1, koff = (tid & 1) << 5;
      __builtin_prefetch(
          &x[(size_t)(rowG0 + r) * D_MODEL + kc + 64 + koff], 0, 3);
    }

    #pragma unroll
    for (int kk = 0; kk < 2; ++kk) {
      int k0 = kk * 32;
      BF16Frag Bhi, Blo;
      int n = 16 * wave + (lane & 15);
      load_B_glb(wcat_hi, n, D_MODEL, kc + k0, lane, Bhi);
      load_B_glb(wcat_lo, n, D_MODEL, kc + k0, lane, Blo);
      #pragma unroll
      for (int m = 0; m < 4; ++m) {
        BF16Frag Ahi, Alo;
        load_A_lds(&Xhi[0][0], 64, 16 * m, k0, lane, Ahi);
        load_A_lds(&Xlo[0][0], 64, 16 * m, k0, lane, Alo);
        acc[m] = wmma_bf16(Ahi, Bhi, acc[m]);   // hi*hi
        acc[m] = wmma_bf16(Alo, Bhi, acc[m]);   // lo*hi
        acc[m] = wmma_bf16(Ahi, Blo, acc[m]);   // hi*lo  (~fp32 accuracy)
      }
    }
  }

  // scatter GEMM1 accumulators into LDS (section = dt/phase/Bval)
  {
    int col = 16 * wave + (lane & 15);        // 0..191
    int sec = col >> 6, c = col & 63, grp = lane >> 4;
    #pragma unroll
    for (int m = 0; m < 4; ++m)
      #pragma unroll
      for (int v = 0; v < 8; ++v)
        raw[sec][16 * m + v + 8 * grp][c] = acc[m][v];
  }
  __syncthreads();

  // ================= elementwise state rotate-and-decay =================
  const float PI = 3.14159265358979323846f;
  for (int i = tid; i < ROWS * RANK; i += NTHREADS) {
    int r = i >> 6, j = i & 63;
    int rowG = rowG0 + r;
    float t   = raw[0][r][j] + b_dt[j];
    float dt  = fmaxf(t, 0.0f) + log1pf(__expf(-fabsf(t)));   // softplus
    float ph  = tanhf(raw[1][r][j] + b_ph[j]) * PI;
    float bv  = raw[2][r][j];
    float dec = __expf(-dt * __expf(A_real[j]));
    float ang = dt * A_imag[j] + ph;
    float s, c;
    __sincosf(ang, &s, &c);
    float sre = state_re[(size_t)rowG * RANK + j];
    float sim = state_im[(size_t)rowG * RANK + j];
    float nre = (sre * c - sim * s) * dec + bv;
    float nim = (sre * s + sim * c) * dec;
    new_re[(size_t)rowG * RANK + j] = nre;
    new_im[(size_t)rowG * RANK + j] = nim;
    uint16_t hi = f2bf(nre);
    Rhi[r][j] = hi;
    Rlo[r][j] = f2bf(nre - bf2f(hi));
  }
  __syncthreads();

  // ================= GEMM2: [64 x 64] x [64 x 2048] =====================
  for (int nt = wave; nt < D_MODEL / 16; nt += NWAVES) {
    v8f acc2[4] = {zero, zero, zero, zero};
    #pragma unroll
    for (int kk = 0; kk < 2; ++kk) {
      int k0 = kk * 32;
      BF16Frag Bhi, Blo;
      int n = 16 * nt + (lane & 15);
      load_B_glb(wc_hi, n, RANK, k0, lane, Bhi);
      load_B_glb(wc_lo, n, RANK, k0, lane, Blo);
      #pragma unroll
      for (int m = 0; m < 4; ++m) {
        BF16Frag Ahi, Alo;
        load_A_lds(&Rhi[0][0], RANK, 16 * m, k0, lane, Ahi);
        load_A_lds(&Rlo[0][0], RANK, 16 * m, k0, lane, Alo);
        acc2[m] = wmma_bf16(Ahi, Bhi, acc2[m]);
        acc2[m] = wmma_bf16(Alo, Bhi, acc2[m]);
        acc2[m] = wmma_bf16(Ahi, Blo, acc2[m]);
      }
    }
    int grp = lane >> 4;
    int col = 16 * nt + (lane & 15);
    #pragma unroll
    for (int m = 0; m < 4; ++m)
      #pragma unroll
      for (int v = 0; v < 8; ++v)
        out[(size_t)(rowG0 + 16 * m + v + 8 * grp) * D_MODEL + col] = acc2[m][v];
  }
}

extern "C" void kernel_launch(void* const* d_in, const int* in_sizes, int n_in,
                              void* d_out, int out_size, void* d_ws, size_t ws_size,
                              hipStream_t stream) {
  (void)in_sizes; (void)n_in; (void)out_size; (void)ws_size;
  const float* x        = (const float*)d_in[0];
  const float* state_re = (const float*)d_in[1];
  const float* state_im = (const float*)d_in[2];
  const float* W_dt     = (const float*)d_in[3];
  const float* b_dt     = (const float*)d_in[4];
  const float* W_ph     = (const float*)d_in[5];
  const float* b_ph     = (const float*)d_in[6];
  const float* W_B      = (const float*)d_in[7];
  const float* W_C      = (const float*)d_in[8];
  const float* A_real   = (const float*)d_in[9];
  const float* A_imag   = (const float*)d_in[10];

  // workspace: split weights (hi + residual), ~2 MB total, L2-resident
  uint16_t* wcat_hi = (uint16_t*)d_ws;
  uint16_t* wcat_lo = wcat_hi + (size_t)NCAT * D_MODEL;
  uint16_t* wc_hi   = wcat_lo + (size_t)NCAT * D_MODEL;
  uint16_t* wc_lo   = wc_hi   + (size_t)D_MODEL * RANK;

  float* out    = (float*)d_out;
  float* nre    = out + (size_t)BATCH * D_MODEL;
  float* nim    = nre + (size_t)BATCH * RANK;

  prep_weights<<<(NCAT * D_MODEL + 255) / 256, 256, 0, stream>>>(
      W_dt, W_ph, W_B, W_C, wcat_hi, wcat_lo, wc_hi, wc_lo);

  mamba3_fused<<<BATCH / ROWS, NTHREADS, 0, stream>>>(
      x, state_re, state_im, b_dt, b_ph, A_real, A_imag,
      wcat_hi, wcat_lo, wc_hi, wc_lo, out, nre, nim);
}
